// GNN_61040075210810
// MI455X (gfx1250) — matrix operand order repaired and checked by reading, not compiled, not run on hardware
//
#include <hip/hip_runtime.h>
#include <hip/hip_bf16.h>

typedef __attribute__((ext_vector_type(16))) _Float16 v16h;
typedef __attribute__((ext_vector_type(8)))  float    v8f;
typedef __attribute__((ext_vector_type(2)))  float    v2f;

union AFrag { v16h v; _Float16 e[16]; };
union CFrag { v8f  v; float    e[8];  };
union A2Frag { v2f v; float    e[2];  };

#define NEGBIG -1000000000.0f

#if defined(__has_builtin)
#if __has_builtin(__builtin_amdgcn_wmma_f32_16x16x4_f32)
#define HAVE_WMMA_F32K4 1
#endif
#endif

// ---------------------------------------------------------------------------
// Pack weight matrix W[128][O] (f32) into per-lane WMMA B-fragment layout f16:
//   Wp[((tn*4 + kt)*32 + lane)*16 + slot] = (f16) W[(slot + (lane>=16?16:0) + kt*32)*O + tn*16 + (lane&15)]
// so each lane's 32x16 f16 B fragment is 16 contiguous halfs (32B).
// ---------------------------------------------------------------------------
__global__ void wpack_kernel(const float* __restrict__ W,
                             _Float16* __restrict__ Wp, int O) {
    int total = O * 128;                       // NT*4*32*16
    int idx = blockIdx.x * 256 + threadIdx.x;
    if (idx >= total) return;
    int slot = idx & 15;
    int lane = (idx >> 4) & 31;
    int kt   = (idx >> 9) & 3;
    int tn   = idx >> 11;
    int k = slot + ((lane >> 4) ? 16 : 0) + kt * 32;
    int n = tn * 16 + (lane & 15);
    Wp[idx] = (_Float16)W[k * O + n];
}

// ---------------------------------------------------------------------------
// Embedding: build fin features, h = fin @ W_emb + b_emb (f16 out), imp mask
// ---------------------------------------------------------------------------
__global__ __launch_bounds__(256)
void embed_kernel(const float* __restrict__ team_obs,    // [B][16][14]
                  const float* __restrict__ target_obs,  // [B][48][15]
                  const float* __restrict__ team_mask,   // [B][2][16]
                  const float* __restrict__ ltm,         // [B][49]
                  const float* __restrict__ W_emb,       // [16][128]
                  const float* __restrict__ b_emb,       // [128]
                  _Float16* __restrict__ x0,             // [B][64][128]
                  unsigned char* __restrict__ imp)       // [B][64]
{
    const int b   = blockIdx.x;
    const int tid = threadIdx.x;
    __shared__ float Wsh[16 * 128];
    __shared__ float fin[64 * 16];

    for (int i = tid; i < 16 * 128; i += 256) Wsh[i] = W_emb[i];

    for (int i = tid; i < 64 * 16; i += 256) {
        int m = i >> 4, k = i & 15;
        float v = 0.0f;
        if (m < 16) {
            if (k < 14) v = team_obs[(b * 16 + m) * 14 + k];
        } else {
            int t = m - 16;
            if (k < 12)       v = target_obs[(b * 48 + t) * 15 + k];
            else if (k >= 14) v = target_obs[(b * 48 + t) * 15 + (k - 2)];
        }
        fin[i] = v;
    }
    if (tid < 64) {
        bool v;
        if (tid < 16) v = (team_mask[(b * 2 + 0) * 16 + tid] != NEGBIG);
        else          v = (ltm[b * 49 + (tid - 16)] != 0.0f);
        imp[b * 64 + tid] = v ? 1 : 0;
    }
    __syncthreads();

    for (int el = tid; el < 64 * 128; el += 256) {
        int m = el >> 7, f = el & 127;
        float acc = b_emb[f];
#pragma unroll
        for (int k = 0; k < 16; ++k) acc += fin[m * 16 + k] * Wsh[k * 128 + f];
        x0[(size_t)b * 64 * 128 + el] = (_Float16)acc;
    }
}

// ---------------------------------------------------------------------------
// Fused GATv2 layer. One block per batch element; 8 wave32s.
//   xl/xr : v_wmma_f32_16x16x32_f16 (pre-packed fragments, b128 loads)
//   aggregation : v_wmma_f32_16x16x4_f32 (full f32, softmax weights)
// ---------------------------------------------------------------------------
template <int H, int C, bool LN, bool FINAL>
__global__ __launch_bounds__(256)
void gat_kernel(const _Float16* __restrict__ x_in,  // [B][64][128] f16 row-major
                const _Float16* __restrict__ Wlp,   // packed B-frags
                const _Float16* __restrict__ Wrp,   // packed B-frags
                const float* __restrict__ att,      // [H*C]
                const float* __restrict__ bias,     // [O]
                const float* __restrict__ ln_g,     // [O] (LN only)
                const float* __restrict__ ln_b,     // [O] (LN only)
                const unsigned char* __restrict__ imp, // [B][64]
                _Float16* __restrict__ x_out,       // [B][64][O] f16 (next)
                float* __restrict__ y_out)          // [B][64][O] f32 (final)
{
    constexpr int O   = H * C;
    constexpr int NT  = O / 16;     // N tiles per GEMM
    constexpr int M   = 64;
    constexpr int F   = 128;
    constexpr int JH  = M * H;      // attention columns
    constexpr int XLS = O + 1;      // padded row stride (f32) for xl/xr

    extern __shared__ __align__(16) char smem[];
    _Float16* apack = (_Float16*)smem;                         // [4*4*32*16] f16
    float* xl = (float*)(smem + M * F * sizeof(_Float16));     // [64][XLS]
    float* xr = xl + M * XLS;                                  // [64][XLS]
    float* esh = xr + M * XLS;                                 // [64][JH] alpha
    unsigned char* impsh = (unsigned char*)(esh + M * JH);     // [64]

    __shared__ float mu_sh[64];
    __shared__ float rs_sh[64];

    const int b    = blockIdx.x;
    const int tid  = threadIdx.x;
    const int lane = tid & 31;
    const int wave = tid >> 5;
    const int hi   = lane >> 4;
    const int lrow = lane & 15;

    // ---- stage x into LDS, scattered into A-fragment layout -------------
    // A (16x32 f16): lanes0-15 K=0-15, lanes16-31 K=16-31
    {
        const unsigned int* xg = (const unsigned int*)(x_in + (size_t)b * M * F);
        unsigned int* ap = (unsigned int*)apack;
        for (int i = tid; i < M * F / 2; i += 256) {
            unsigned int d = xg[i];          // halfs (m, 2q) and (m, 2q+1)
            int m = i >> 6;
            int k = (i & 63) * 2;
            int tm = m >> 4, lr = m & 15;
            int kt = k >> 5, kk = k & 31;
            int hh = kk >> 4;
            int slot = kk & 15;
            int dst = (((tm * 4 + kt) * 32 + hh * 16 + lr) << 4) + slot;  // halfs
            ap[dst >> 1] = d;
        }
    }
    if (tid < 64) impsh[tid] = imp[b * 64 + tid];
    __syncthreads();

    // ---------------- WMMA GEMMs: xl = x@Wl, xr = x@Wr -------------------
    {
        const _Float16* Wp = (wave >= 4) ? Wrp : Wlp;
        float*          xo = (wave >= 4) ? xr : xl;
        const int tm = wave & 3;
        for (int tn = 0; tn < NT; ++tn) {
            CFrag c = {};
#pragma unroll
            for (int kt = 0; kt < 4; ++kt) {
                AFrag a, bb;
                a.v  = *(const v16h*)(apack + ((tm * 4 + kt) * 32 + lane) * 16);
                bb.v = *(const v16h*)(Wp + (size_t)((tn * 4 + kt) * 32 + lane) * 16);
                c.v = __builtin_amdgcn_wmma_f32_16x16x32_f16(
                    false, a.v, false, bb.v, (short)0, c.v, false, false);
            }
#pragma unroll
            for (int r = 0; r < 8; ++r) {
                int mm = tm * 16 + r + (hi ? 8 : 0);
                xo[mm * XLS + tn * 16 + lrow] = c.e[r];
            }
        }
    }
    __syncthreads();

    // ---------------- attention scores + masked softmax ------------------
    const int jn = tid / H;
    const int hn = tid % H;
    if (tid < JH) {
        const bool impj = (impsh[jn] != 0);
        float maxv = -3.0e38f;
        for (int i = 0; i < M; ++i) {
            bool adj = (i == jn) || (impj && (impsh[i] != 0));
            float ev;
            if (adj) {
                ev = 0.0f;
#pragma unroll 8
                for (int c = 0; c < C; ++c) {
                    float z = xl[i * XLS + hn * C + c] + xr[jn * XLS + hn * C + c];
                    z = (z > 0.0f) ? z : 0.2f * z;
                    ev += att[hn * C + c] * z;
                }
            } else {
                ev = NEGBIG;
            }
            esh[i * JH + tid] = ev;
            maxv = fmaxf(maxv, ev);
        }
        float sum = 0.0f;
        for (int i = 0; i < M; ++i) {
            float ex = __expf(esh[i * JH + tid] - maxv);
            esh[i * JH + tid] = ex;
            sum += ex;
        }
        float inv = 1.0f / sum;
        for (int i = 0; i < M; ++i) esh[i * JH + tid] *= inv;   // alpha
    }
    __syncthreads();   // alpha visible to all waves

    // ---------------- aggregation: out = alpha^T @ xl (per head) ---------
    constexpr int NT2 = C / 16;
    constexpr int TT  = H * 4 * NT2;       // output tiles
    constexpr int TPW = TT / 8;            // tiles per wave
    CFrag oc[TPW];

#if HAVE_WMMA_F32K4
    for (int t = 0; t < TPW; ++t) {
        const int tile = wave * TPW + t;
        const int tnc  = tile % NT2;
        const int tj   = (tile / NT2) & 3;
        const int hh   = tile / (NT2 * 4);
        CFrag c = {};
        for (int ks = 0; ks < 16; ++ks) {
            A2Frag a2, b2;
            const int ia = ks * 4 + (hi ? 2 : 0);
            // A = alpha^T: M=j (lanes), K=i
            a2.e[0] = esh[(ia    ) * JH + (tj * 16 + lrow) * H + hh];
            a2.e[1] = esh[(ia + 1) * JH + (tj * 16 + lrow) * H + hh];
            // B = xl_h: K=i, N=c (lanes)
            b2.e[0] = xl[(ia    ) * XLS + hh * C + tnc * 16 + lrow];
            b2.e[1] = xl[(ia + 1) * XLS + hh * C + tnc * 16 + lrow];
            c.v = __builtin_amdgcn_wmma_f32_16x16x4_f32(
                false, a2.v, false, b2.v, (short)0, c.v, false, false);
        }
        oc[t] = c;
    }
#else
    // VALU fallback: each wave still produces its TPW 16x16 tiles
    for (int t = 0; t < TPW; ++t) {
        const int tile = wave * TPW + t;
        const int tnc  = tile % NT2;
        const int tj   = (tile / NT2) & 3;
        const int hh   = tile / (NT2 * 4);
#pragma unroll
        for (int r = 0; r < 8; ++r) {
            int row = tj * 16 + r + (hi ? 8 : 0);
            int col = hh * C + tnc * 16 + lrow;
            float s = 0.0f;
            for (int i = 0; i < M; ++i)
                s += esh[i * JH + row * H + hh] * xl[i * XLS + col];
            oc[t].e[r] = s;
        }
    }
#endif
    __syncthreads();   // all alpha/xl reads done before esh reuse

    // ---------------- epilogue: bias (+LN+ReLU) / final store ------------
    if constexpr (FINAL) {
        for (int t = 0; t < TPW; ++t) {
            const int tile = wave * TPW + t;
            const int tnc  = tile % NT2;
            const int tj   = (tile / NT2) & 3;
            const int hh   = tile / (NT2 * 4);
            const int col  = hh * C + tnc * 16 + lrow;
#pragma unroll
            for (int r = 0; r < 8; ++r) {
                int row = tj * 16 + r + (hi ? 8 : 0);
                y_out[((size_t)b * M + row) * O + col] = oc[t].e[r] + bias[col];
            }
        }
    } else {
        constexpr int OS = O + 2;
        float* osh = esh;   // reuse attention LDS as output staging
        for (int t = 0; t < TPW; ++t) {
            const int tile = wave * TPW + t;
            const int tnc  = tile % NT2;
            const int tj   = (tile / NT2) & 3;
            const int hh   = tile / (NT2 * 4);
            const int col  = hh * C + tnc * 16 + lrow;
#pragma unroll
            for (int r = 0; r < 8; ++r) {
                int row = tj * 16 + r + (hi ? 8 : 0);
                osh[row * OS + col] = oc[t].e[r] + bias[col];
            }
        }
        __syncthreads();
        if constexpr (LN) {
            if (tid < M) {
                float s = 0.0f;
                for (int f = 0; f < O; ++f) s += osh[tid * OS + f];
                float mu = s / (float)O;
                float s2 = 0.0f;
                for (int f = 0; f < O; ++f) {
                    float d = osh[tid * OS + f] - mu;
                    s2 += d * d;
                }
                mu_sh[tid] = mu;
                rs_sh[tid] = rsqrtf(s2 / (float)O + 1e-5f);
            }
            __syncthreads();
            for (int el = tid; el < M * O; el += 256) {
                int jj = el / O, f = el % O;
                float v = (osh[jj * OS + f] - mu_sh[jj]) * rs_sh[jj] * ln_g[f] + ln_b[f];
                v = fmaxf(v, 0.0f);   // ReLU
                x_out[(size_t)b * M * O + el] = (_Float16)v;
            }
        } else {
            for (int el = tid; el < M * O; el += 256)
                x_out[(size_t)b * M * O + el] = (_Float16)osh[(el / O) * OS + (el % O)];
        }
    }
}

// ---------------------------------------------------------------------------
extern "C" void kernel_launch(void* const* d_in, const int* in_sizes, int n_in,
                              void* d_out, int out_size, void* d_ws, size_t ws_size,
                              hipStream_t stream) {
    const float* team_obs   = (const float*)d_in[0];
    const float* target_obs = (const float*)d_in[1];
    const float* team_mask  = (const float*)d_in[2];
    const float* ltm        = (const float*)d_in[3];
    const float* W_emb      = (const float*)d_in[4];
    const float* b_emb      = (const float*)d_in[5];
    const float* Wl1        = (const float*)d_in[6];
    const float* Wr1        = (const float*)d_in[7];
    const float* att1       = (const float*)d_in[8];
    const float* b1         = (const float*)d_in[9];
    const float* Wl2        = (const float*)d_in[10];
    const float* Wr2        = (const float*)d_in[11];
    const float* att2       = (const float*)d_in[12];
    const float* b2         = (const float*)d_in[13];
    const float* Wl3        = (const float*)d_in[14];
    const float* Wr3        = (const float*)d_in[15];
    const float* att3       = (const float*)d_in[16];
    const float* b3         = (const float*)d_in[17];
    const float* ln1_g      = (const float*)d_in[18];
    const float* ln1_b      = (const float*)d_in[19];
    const float* ln2_g      = (const float*)d_in[20];
    const float* ln2_b      = (const float*)d_in[21];

    const int B = 128;

    // ---- workspace layout ----
    char* ws = (char*)d_ws;
    _Float16* Wl1p = (_Float16*)ws;
    _Float16* Wr1p = Wl1p + 128 * 128;
    _Float16* Wl2p = Wr1p + 128 * 128;
    _Float16* Wr2p = Wl2p + 128 * 128;
    _Float16* Wl3p = Wr2p + 128 * 128;
    _Float16* Wr3p = Wl3p + 128 * 64;
    unsigned char* imp = (unsigned char*)(Wr3p + 128 * 64);      // B*64 bytes
    _Float16* x0 = (_Float16*)(imp + B * 64);
    _Float16* x1 = x0 + (size_t)B * 64 * 128;
    _Float16* x2 = x1 + (size_t)B * 64 * 128;

    // ---- weight pack (f32 -> fragment-layout f16) ----
    wpack_kernel<<<(128 * 128 + 255) / 256, 256, 0, stream>>>(Wl1, Wl1p, 128);
    wpack_kernel<<<(128 * 128 + 255) / 256, 256, 0, stream>>>(Wr1, Wr1p, 128);
    wpack_kernel<<<(128 * 128 + 255) / 256, 256, 0, stream>>>(Wl2, Wl2p, 128);
    wpack_kernel<<<(128 * 128 + 255) / 256, 256, 0, stream>>>(Wr2, Wr2p, 128);
    wpack_kernel<<<(128 * 64 + 255) / 256, 256, 0, stream>>>(Wl3, Wl3p, 64);
    wpack_kernel<<<(128 * 64 + 255) / 256, 256, 0, stream>>>(Wr3, Wr3p, 64);

    // ---- embedding ----
    embed_kernel<<<B, 256, 0, stream>>>(team_obs, target_obs, team_mask, ltm,
                                        W_emb, b_emb, x0, imp);

    // ---- dynamic LDS sizes ----
    const size_t smem12 = (size_t)64 * 128 * 2
                        + 2 * (size_t)64 * (128 + 1) * 4
                        + (size_t)64 * (64 * 4) * 4
                        + 64;
    const size_t smem3  = (size_t)64 * 128 * 2
                        + 2 * (size_t)64 * (64 + 1) * 4
                        + (size_t)64 * 64 * 4
                        + 64;

    // ---- GAT layers ----
    gat_kernel<4, 32, true,  false><<<B, 256, smem12, stream>>>(
        x0, Wl1p, Wr1p, att1, b1, ln1_g, ln1_b, imp, x1, nullptr);
    gat_kernel<4, 32, true,  false><<<B, 256, smem12, stream>>>(
        x1, Wl2p, Wr2p, att2, b2, ln2_g, ln2_b, imp, x2, nullptr);
    gat_kernel<1, 64, false, true ><<<B, 256, smem3,  stream>>>(
        x2, Wl3p, Wr3p, att3, b3, nullptr, nullptr, imp, nullptr, (float*)d_out);

    (void)in_sizes; (void)n_in; (void)out_size; (void)ws_size;
}
